// NoiseNet_15135464751286
// MI455X (gfx1250) — compile-verified, matrix operand order checked by reading, not run on hardware
//
#include <hip/hip_runtime.h>
#include <math.h>

#define B_   8
#define P_   12
#define N_   5000
#define H_   8
#define E_   50000
#define NB_  8
#define TD0_ 16
#define TD1_ 64
#define NH_  (N_ * H_)          /* 40000 */
#define BP_  (B_ * P_)          /* 96 */
#define BPN_ (B_ * P_ * N_)     /* 480000 */
#define BPNH_ ((size_t)B_ * P_ * N_ * H_)  /* 3,840,000 */

/* workspace layout (in floats) */
#define H_OFF    ((size_t)0)
#define A_OFF    ((size_t)3840000)
#define G_OFF    ((size_t)7680000)
#define ACC_OFF  ((size_t)11520000)
#define EN_OFF   ((size_t)15360000)
#define DEG_OFF  ((size_t)15410000)
#define DINV_OFF ((size_t)15415000)
#define PP_OFF   ((size_t)15420000)
#define TE_OFF   ((size_t)15460000)
#define TA_OFF   ((size_t)15460512)
#define AS_OFF   ((size_t)15461024)

typedef __attribute__((ext_vector_type(2)))  float    v2f;
typedef __attribute__((ext_vector_type(8)))  float    v8f;
typedef __attribute__((ext_vector_type(16))) _Float16 v16h;

#if defined(__has_builtin)
#if __has_builtin(__builtin_amdgcn_wmma_f32_16x16x4_f32)
#define USE_WMMA_F32X4 1
#endif
#endif

/* ---------- tiny prep: time-embedding MLP, per-block time bias, softmax ---------- */
__global__ void k_prep(const int* __restrict__ time,
                       const float* __restrict__ te_w1, const float* __restrict__ te_b1,
                       const float* __restrict__ te_w2, const float* __restrict__ te_b2,
                       const float* __restrict__ bt_w1, const float* __restrict__ bt_b1,
                       const float* __restrict__ bt_w2, const float* __restrict__ bt_b2,
                       const float* __restrict__ attn,
                       float* __restrict__ te_out, float* __restrict__ t_all,
                       float* __restrict__ a_soft) {
  __shared__ float emb[2 * TD0_];
  __shared__ float l1[TD1_];
  __shared__ float l2[TD1_];
  __shared__ float hid[H_];
  int t = threadIdx.x;  /* 64 threads */
  for (int b = 0; b < B_; ++b) {
    float step = (float)time[b];
    if (t < 2 * TD0_) {
      int d = t % TD0_;
      float e = step * powf(10.0f, (float)d * 4.0f / (float)TD0_);
      emb[t] = (t < TD0_) ? sinf(e) : cosf(e);
    }
    __syncthreads();
    {
      float s = te_b1[t];
      for (int j = 0; j < 2 * TD0_; ++j) s += emb[j] * te_w1[j * TD1_ + t];
      l1[t] = s / (1.0f + expf(-s));  /* silu */
    }
    __syncthreads();
    {
      float s = te_b2[t];
      for (int j = 0; j < TD1_; ++j) s += l1[j] * te_w2[j * TD1_ + t];
      l2[t] = s / (1.0f + expf(-s));
    }
    __syncthreads();
    te_out[b * TD1_ + t] = l2[t];
    for (int i = 0; i < NB_; ++i) {
      if (t < H_) {
        float s = bt_b1[i * H_ + t];
        for (int k = 0; k < TD1_; ++k) s += l2[k] * bt_w1[((size_t)i * TD1_ + k) * H_ + t];
        hid[t] = fmaxf(s, 0.0f);
      }
      __syncthreads();
      if (t < H_) {
        float s = bt_b2[i * H_ + t];
        for (int j = 0; j < H_; ++j) s += hid[j] * bt_w2[((size_t)i * H_ + j) * H_ + t];
        t_all[((size_t)i * B_ + b) * H_ + t] = s;
      }
      __syncthreads();
    }
    __syncthreads();
  }
  if (t == 0) {
    float av[NB_ + 1], mx = -1e30f, sum = 0.0f;
    for (int j = 0; j <= NB_; ++j) { av[j] = attn[j]; mx = fmaxf(mx, av[j]); }
    for (int j = 0; j <= NB_; ++j) { av[j] = expf(av[j] - mx); sum += av[j]; }
    for (int j = 0; j <= NB_; ++j) a_soft[j] = av[j] / sum;
  }
}

/* ---------- GCN norm ---------- */
__global__ void k_deg(const int* __restrict__ dst, float* __restrict__ deg) {
  int e = blockIdx.x * blockDim.x + threadIdx.x;
  if (e < E_) atomicAdd(&deg[dst[e]], 1.0f);
}
__global__ void k_dinv(const float* __restrict__ deg, float* __restrict__ dinv) {
  int n = blockIdx.x * blockDim.x + threadIdx.x;
  if (n < N_) { float d = deg[n]; dinv[n] = (d > 0.0f) ? rsqrtf(d) : 0.0f; }
}
__global__ void k_enorm(const float* __restrict__ dinv, const int* __restrict__ src,
                        const int* __restrict__ dst, float* __restrict__ en) {
  int e = blockIdx.x * blockDim.x + threadIdx.x;
  if (e < E_) en[e] = dinv[src[e]] * dinv[dst[e]];
}

/* ---------- positional projection (per node) ---------- */
__global__ void k_posproj(const float* __restrict__ pos,
                          const float* __restrict__ w1, const float* __restrict__ b1,
                          const float* __restrict__ w2, const float* __restrict__ b2,
                          float* __restrict__ pp) {
  int n = blockIdx.x * blockDim.x + threadIdx.x;
  if (n >= N_) return;
  const float* p = pos + (size_t)n * H_;
  float hid[H_];
#pragma unroll
  for (int j = 0; j < H_; ++j) {
    float s = b1[j];
#pragma unroll
    for (int k = 0; k < H_; ++k) s += p[k] * w1[k * H_ + j];
    hid[j] = fmaxf(s, 0.0f);
  }
#pragma unroll
  for (int h = 0; h < H_; ++h) {
    float s = b2[h];
#pragma unroll
    for (int j = 0; j < H_; ++j) s += hid[j] * w2[j * H_ + h];
    pp[(size_t)n * H_ + h] = s;
  }
}

/* ---------- input projection + pos, seed skip accumulator ---------- */
__global__ void k_init(const float* __restrict__ x,
                       const float* __restrict__ w1, const float* __restrict__ b1,
                       const float* __restrict__ w2, const float* __restrict__ b2,
                       const float* __restrict__ pp, const float* __restrict__ a_soft,
                       float* __restrict__ hbuf, float* __restrict__ acc) {
  int idx = blockIdx.x * blockDim.x + threadIdx.x;
  if (idx >= BPN_) return;
  int n = idx % N_;
  float xv = x[idx];
  float hid[H_];
#pragma unroll
  for (int j = 0; j < H_; ++j) hid[j] = fmaxf(xv * w1[j] + b1[j], 0.0f);
  float a0 = a_soft[0];
#pragma unroll
  for (int h = 0; h < H_; ++h) {
    float s = b2[h];
#pragma unroll
    for (int j = 0; j < H_; ++j) s += hid[j] * w2[j * H_ + h];
    s += pp[(size_t)n * H_ + h];
    hbuf[(size_t)idx * H_ + h] = s;
    acc[(size_t)idx * H_ + h] = a0 * s;
  }
}

/* ---------- fused cond-proj + add + LayerNorm over (N,H) ---------- */
__global__ void __launch_bounds__(256)
k_ln1(const float* __restrict__ hbuf, const float* __restrict__ cond,
      const float* __restrict__ w1, const float* __restrict__ b1,
      const float* __restrict__ w2, const float* __restrict__ b2,
      const float* __restrict__ gw, const float* __restrict__ gb,
      float* __restrict__ out) {
  int bp = blockIdx.x, t = threadIdx.x;
  __shared__ float sw1[64], sb1[8], sw2[64], sb2[8];
  __shared__ float red[256], red2[256];
  __shared__ float mean_s, rstd_s;
  if (t < 64) { sw1[t] = w1[t]; sw2[t] = w2[t]; }
  if (t < 8)  { sb1[t] = b1[t]; sb2[t] = b2[t]; }
  __syncthreads();
  const float* hb = hbuf + (size_t)bp * NH_;
  const float* cd = cond + (size_t)bp * NH_;
  float* ob = out + (size_t)bp * NH_;
  float sum = 0.0f, ssq = 0.0f;
  for (int n = t; n < N_; n += 256) {
    const float* cp = cd + (size_t)n * H_;
    float hid[H_];
#pragma unroll
    for (int j = 0; j < H_; ++j) {
      float s = sb1[j];
#pragma unroll
      for (int k = 0; k < H_; ++k) s += cp[k] * sw1[k * H_ + j];
      hid[j] = fmaxf(s, 0.0f);
    }
#pragma unroll
    for (int h = 0; h < H_; ++h) {
      float s = sb2[h];
#pragma unroll
      for (int j = 0; j < H_; ++j) s += hid[j] * sw2[j * H_ + h];
      float v = hb[(size_t)n * H_ + h] + s;
      ob[(size_t)n * H_ + h] = v;
      sum += v; ssq += v * v;
    }
  }
  red[t] = sum; red2[t] = ssq;
  __syncthreads();
  for (int s = 128; s > 0; s >>= 1) {
    if (t < s) { red[t] += red[t + s]; red2[t] += red2[t + s]; }
    __syncthreads();
  }
  if (t == 0) {
    float m = red[0] / (float)NH_;
    float v = red2[0] / (float)NH_ - m * m;
    mean_s = m; rstd_s = rsqrtf(fmaxf(v, 0.0f) + 1e-5f);
  }
  __syncthreads();
  float m = mean_s, r = rstd_s;
  for (int nh = t; nh < NH_; nh += 256)
    ob[nh] = (ob[nh] - m) * r * gw[nh] + gb[nh];
}

/* ---------- graph conv: scatter-add per edge ---------- */
__global__ void k_gconv(const float* __restrict__ in, const int* __restrict__ src,
                        const int* __restrict__ dst, const float* __restrict__ en,
                        float* __restrict__ out) {
  int e = blockIdx.x * blockDim.x + threadIdx.x;
  if (e >= E_) return;
  int bp = blockIdx.y;
  float w = en[e];
  const float* ip = in + (size_t)bp * NH_ + (size_t)src[e] * H_;
  float* op = out + (size_t)bp * NH_ + (size_t)dst[e] * H_;
#pragma unroll
  for (int h = 0; h < H_; ++h) atomicAdd(&op[h], w * ip[h]);
}

/* ---------- fused time-bias add + LayerNorm ---------- */
__global__ void __launch_bounds__(256)
k_ln2(const float* __restrict__ gbuf, const float* __restrict__ t_all_i,
      const float* __restrict__ tw, const float* __restrict__ tb,
      float* __restrict__ out) {
  int bp = blockIdx.x, t = threadIdx.x;
  int b = bp / P_;
  __shared__ float tv[H_];
  __shared__ float red[256], red2[256];
  __shared__ float mean_s, rstd_s;
  if (t < H_) tv[t] = t_all_i[b * H_ + t];
  __syncthreads();
  const float* gp = gbuf + (size_t)bp * NH_;
  float* op = out + (size_t)bp * NH_;
  float sum = 0.0f, ssq = 0.0f;
  for (int nh = t; nh < NH_; nh += 256) {
    float v = gp[nh] + tv[nh & (H_ - 1)];
    op[nh] = v; sum += v; ssq += v * v;
  }
  red[t] = sum; red2[t] = ssq;
  __syncthreads();
  for (int s = 128; s > 0; s >>= 1) {
    if (t < s) { red[t] += red[t + s]; red2[t] += red2[t + s]; }
    __syncthreads();
  }
  if (t == 0) {
    float m = red[0] / (float)NH_;
    float v = red2[0] / (float)NH_ - m * m;
    mean_s = m; rstd_s = rsqrtf(fmaxf(v, 0.0f) + 1e-5f);
  }
  __syncthreads();
  float m = mean_s, r = rstd_s;
  for (int nh = t; nh < NH_; nh += 256)
    op[nh] = (op[nh] - m) * r * tw[nh] + tb[nh];
}

/* ---------- P-conv as WMMA GEMM: out[q,col] = sum_p cv[q,p]*in[p,col] ----------
   M=12(pad16), K=12 (3 chunks of 4 for f32 WMMA), Ncols = NH_ in tiles of 16.
   cv_w is staged (zero-padded) in LDS once per block; A operands are built
   branchlessly from LDS and reused across TPW_ tiles per wave.
   Also writes new state h and accumulates a[i+1]*h into skip accumulator.   */
#define TPW_    5                        /* tiles per wave */
#define TILES_  (NH_ / 16)               /* 2500 */
#define GROUPS_ (TILES_ / TPW_)          /* 500 tile-groups per batch */

__global__ void __launch_bounds__(256)
k_convp(const float* __restrict__ in, const float* __restrict__ cvw,
        const float* __restrict__ cvb, const float* __restrict__ a_soft,
        int blk_i, float* __restrict__ hout, float* __restrict__ acc) {
  __shared__ float scv[16 * P_];   /* A matrix, rows 12..15 zero-padded */
  __shared__ float scb[16];        /* bias, entries 12..15 zero          */
  int t = threadIdx.x;
  if (t < 16 * P_) scv[t] = (t < P_ * P_) ? cvw[t] : 0.0f;
  if (t < 16)      scb[t] = (t < P_) ? cvb[t] : 0.0f;
  __syncthreads();

  int wave = t >> 5;
  int lane = t & 31;
  int wj = blockIdx.x * 8 + wave;   /* wave-job id: 0 .. B_*GROUPS_-1 */
  int b = wj / GROUPS_;             /* 0..7, exact */
  int tile0 = (wj % GROUPS_) * TPW_;
  int half = lane >> 4;             /* 0: lanes 0-15, 1: lanes 16-31 */
  int m = lane & 15;
  const float* ib = in + (size_t)b * P_ * NH_;
  float* hb = hout + (size_t)b * P_ * NH_;
  float* ab = acc + (size_t)b * P_ * NH_;
  float aw = a_soft[blk_i + 1];

#ifdef USE_WMMA_F32X4
  /* A operands: 16x4 f32 layout -> VGPR0: K = ka (lanes0-15) / ka (lanes16-31
     with +2), VGPR1: K = ka+1 / ka+3.  Built once, reused for all tiles. */
  v2f a0, a1, a2;
  {
    int k0 = 2 * half;              /* chunk base offsets 0,4,8 */
    a0.x = scv[m * P_ + k0];     a0.y = scv[m * P_ + k0 + 1];
    a1.x = scv[m * P_ + 4 + k0]; a1.y = scv[m * P_ + 4 + k0 + 1];
    a2.x = scv[m * P_ + 8 + k0]; a2.y = scv[m * P_ + 8 + k0 + 1];
  }
#pragma unroll
  for (int tt = 0; tt < TPW_; ++tt) {
    int col = (tile0 + tt) * 16 + m;
    int kb = 2 * half;
    v2f b0, b1, b2;
    b0.x = ib[(size_t)(kb)*NH_ + col];      b0.y = ib[(size_t)(kb + 1) * NH_ + col];
    b1.x = ib[(size_t)(4 + kb) * NH_ + col]; b1.y = ib[(size_t)(5 + kb) * NH_ + col];
    b2.x = ib[(size_t)(8 + kb) * NH_ + col]; b2.y = ib[(size_t)(9 + kb) * NH_ + col];
    v8f c = {0.f, 0.f, 0.f, 0.f, 0.f, 0.f, 0.f, 0.f};
    c = __builtin_amdgcn_wmma_f32_16x16x4_f32(false, a0, false, b0, (short)0, c, false, false);
    c = __builtin_amdgcn_wmma_f32_16x16x4_f32(false, a1, false, b1, (short)0, c, false, false);
    c = __builtin_amdgcn_wmma_f32_16x16x4_f32(false, a2, false, b2, (short)0, c, false, false);
#pragma unroll
    for (int r = 0; r < 8; ++r) {
      int q = r + 8 * half;         /* C/D layout: VGPR r -> M=r / M=r+8 */
      if (q < P_) {
        float v = c[r] + scb[q];
        hb[(size_t)q * NH_ + col] = v;
        ab[(size_t)q * NH_ + col] += aw * v;
      }
    }
  }
#else
  /* fallback: codegen-confirmed f16 WMMA, K=12 padded into K=32 */
  v16h af;
#pragma unroll
  for (int r = 0; r < 8; ++r) {
    int kbase = (r < 4) ? (2 * r + 8 * half) : (16 + 2 * (r - 4) + 8 * half);
    af[2 * r]     = (_Float16)((kbase < P_) ? scv[m * P_ + kbase] : 0.0f);
    af[2 * r + 1] = (_Float16)((kbase + 1 < P_) ? scv[m * P_ + kbase + 1] : 0.0f);
  }
#pragma unroll
  for (int tt = 0; tt < TPW_; ++tt) {
    int col = (tile0 + tt) * 16 + m;
    v16h bf;
#pragma unroll
    for (int r = 0; r < 8; ++r) {
      int kbase = (r < 4) ? (2 * r + 8 * half) : (16 + 2 * (r - 4) + 8 * half);
      bf[2 * r]     = (_Float16)((kbase < P_) ? ib[(size_t)kbase * NH_ + col] : 0.0f);
      bf[2 * r + 1] = (_Float16)((kbase + 1 < P_) ? ib[(size_t)(kbase + 1) * NH_ + col] : 0.0f);
    }
    v8f c = {0.f, 0.f, 0.f, 0.f, 0.f, 0.f, 0.f, 0.f};
    c = __builtin_amdgcn_wmma_f32_16x16x32_f16(false, af, false, bf, (short)0, c, false, false);
#pragma unroll
    for (int r = 0; r < 8; ++r) {
      int q = r + 8 * half;
      if (q < P_) {
        float v = c[r] + scb[q];
        hb[(size_t)q * NH_ + col] = v;
        ab[(size_t)q * NH_ + col] += aw * v;
      }
    }
  }
#endif
}

/* ---------- final projection ---------- */
__global__ void k_out(const float* __restrict__ acc,
                      const float* __restrict__ w1, const float* __restrict__ b1,
                      const float* __restrict__ w2, const float* __restrict__ b2,
                      float* __restrict__ out) {
  int idx = blockIdx.x * blockDim.x + threadIdx.x;
  if (idx >= BPN_) return;
  const float* ap = acc + (size_t)idx * H_;
  float hid[H_];
#pragma unroll
  for (int j = 0; j < H_; ++j) {
    float s = b1[j];
#pragma unroll
    for (int h = 0; h < H_; ++h) s += ap[h] * w1[h * H_ + j];
    hid[j] = fmaxf(s, 0.0f);
  }
  float o = b2[0];
#pragma unroll
  for (int j = 0; j < H_; ++j) o += hid[j] * w2[j];
  out[idx] = o;
}

extern "C" void kernel_launch(void* const* d_in, const int* in_sizes, int n_in,
                              void* d_out, int out_size, void* d_ws, size_t ws_size,
                              hipStream_t stream) {
  const float* x      = (const float*)d_in[0];
  const float* cond   = (const float*)d_in[1];
  const float* pos    = (const float*)d_in[2];
  const int*   time   = (const int*)d_in[3];
  const int*   ei     = (const int*)d_in[4];
  const float* te_w1  = (const float*)d_in[5];
  const float* te_b1  = (const float*)d_in[6];
  const float* te_w2  = (const float*)d_in[7];
  const float* te_b2  = (const float*)d_in[8];
  const float* in_w1  = (const float*)d_in[9];
  const float* in_b1  = (const float*)d_in[10];
  const float* in_w2  = (const float*)d_in[11];
  const float* in_b2  = (const float*)d_in[12];
  const float* pos_w1 = (const float*)d_in[13];
  const float* pos_b1 = (const float*)d_in[14];
  const float* pos_w2 = (const float*)d_in[15];
  const float* pos_b2 = (const float*)d_in[16];
  const float* attn   = (const float*)d_in[17];
  const float* bt_w1  = (const float*)d_in[18];
  const float* bt_b1  = (const float*)d_in[19];
  const float* bt_w2  = (const float*)d_in[20];
  const float* bt_b2  = (const float*)d_in[21];
  const float* bc_w1  = (const float*)d_in[22];
  const float* bc_b1  = (const float*)d_in[23];
  const float* bc_w2  = (const float*)d_in[24];
  const float* bc_b2  = (const float*)d_in[25];
  const float* gn_w   = (const float*)d_in[26];
  const float* gn_b   = (const float*)d_in[27];
  const float* tn_w   = (const float*)d_in[28];
  const float* tn_b   = (const float*)d_in[29];
  const float* cv_w   = (const float*)d_in[30];
  const float* cv_b   = (const float*)d_in[31];
  const float* out_w1 = (const float*)d_in[32];
  const float* out_b1 = (const float*)d_in[33];
  const float* out_w2 = (const float*)d_in[34];
  const float* out_b2 = (const float*)d_in[35];

  float* ws     = (float*)d_ws;
  float* hbuf   = ws + H_OFF;
  float* bufA   = ws + A_OFF;
  float* buf2   = ws + G_OFF;
  float* acc    = ws + ACC_OFF;
  float* enorm  = ws + EN_OFF;
  float* deg    = ws + DEG_OFF;
  float* dinv   = ws + DINV_OFF;
  float* pp     = ws + PP_OFF;
  float* te2    = ws + TE_OFF;
  float* t_all  = ws + TA_OFF;
  float* a_soft = ws + AS_OFF;

  const int* src = ei;
  const int* dst = ei + E_;

  hipMemsetAsync(deg, 0, N_ * sizeof(float), stream);
  k_prep<<<1, 64, 0, stream>>>(time, te_w1, te_b1, te_w2, te_b2,
                               bt_w1, bt_b1, bt_w2, bt_b2, attn,
                               te2, t_all, a_soft);
  k_deg<<<(E_ + 255) / 256, 256, 0, stream>>>(dst, deg);
  k_dinv<<<(N_ + 255) / 256, 256, 0, stream>>>(deg, dinv);
  k_enorm<<<(E_ + 255) / 256, 256, 0, stream>>>(dinv, src, dst, enorm);
  k_posproj<<<(N_ + 255) / 256, 256, 0, stream>>>(pos, pos_w1, pos_b1, pos_w2, pos_b2, pp);
  k_init<<<(BPN_ + 255) / 256, 256, 0, stream>>>(x, in_w1, in_b1, in_w2, in_b2,
                                                 pp, a_soft, hbuf, acc);

  for (int i = 0; i < NB_; ++i) {
    k_ln1<<<BP_, 256, 0, stream>>>(hbuf, cond,
                                   bc_w1 + (size_t)i * 64, bc_b1 + (size_t)i * 8,
                                   bc_w2 + (size_t)i * 64, bc_b2 + (size_t)i * 8,
                                   gn_w + (size_t)i * NH_, gn_b + (size_t)i * NH_,
                                   bufA);
    hipMemsetAsync(buf2, 0, BPNH_ * sizeof(float), stream);
    dim3 gg((E_ + 255) / 256, BP_);
    k_gconv<<<gg, 256, 0, stream>>>(bufA, src, dst, enorm, buf2);
    k_ln2<<<BP_, 256, 0, stream>>>(buf2, t_all + (size_t)i * B_ * H_,
                                   tn_w + (size_t)i * NH_, tn_b + (size_t)i * NH_,
                                   bufA);
    /* 8 waves/block, TPW_ tiles/wave: B_*GROUPS_/8 = 500 blocks */
    k_convp<<<(B_ * GROUPS_) / 8, 256, 0, stream>>>(bufA,
                                   cv_w + (size_t)i * P_ * P_, cv_b + (size_t)i * P_,
                                   a_soft, i, hbuf, acc);
  }

  k_out<<<(BPN_ + 255) / 256, 256, 0, stream>>>(acc, out_w1, out_b1, out_w2, out_b2,
                                                (float*)d_out);
}
